// TibedoLayer_84198538871241
// MI455X (gfx1250) — matrix operand order repaired and checked by reading, not compiled, not run on hardware
//
#include <hip/hip_runtime.h>
#include <hip/hip_bf16.h>
#include <math.h>

#define B_   8192
#define IN_  4096
#define OUT_ 4096
#define P_   100
#define Q_   7

typedef __attribute__((ext_vector_type(16))) __bf16 v16bf;
typedef __attribute__((ext_vector_type(8)))  float  v8f;

union Frag { v16bf bf; uint4 u[2]; };

__device__ __forceinline__ unsigned short f2bf(float f) {
    unsigned int u = __float_as_uint(f);
    u += ((u >> 16) & 1u) + 0x7FFFu;   // round-to-nearest-even
    return (unsigned short)(u >> 16);
}

// ---- cosine weights wn[p] = cos(pi*p/99) / sum ----
__global__ void wn_kernel(float* __restrict__ wn) {
    __shared__ float sh[128];
    int t = threadIdx.x;
    float c = 0.0f;
    if (t < P_) c = cosf(3.14159265358979323846f * (float)t / (float)(P_ - 1));
    sh[t] = c;
    __syncthreads();
    for (int off = 64; off > 0; off >>= 1) {
        if (t < off) sh[t] += sh[t + off];
        __syncthreads();
    }
    if (t < P_) wn[t] = c / sh[0];
}

// ---- qs[o] = 0.1 * sum_q QT[o,q] ----
__global__ void qs_kernel(const float* __restrict__ qt, float* __restrict__ qs) {
    int o = blockIdx.x * blockDim.x + threadIdx.x;
    if (o < OUT_) {
        float a = 0.0f;
        #pragma unroll
        for (int q = 0; q < Q_; q++) a += qt[o * Q_ + q];
        qs[o] = 0.1f * a;
    }
}

// ---- s[b] = sum_{p<100} x[b,p] * wn[p] ----
__global__ void s_kernel(const float* __restrict__ x, const float* __restrict__ wn,
                         float* __restrict__ s) {
    int b = blockIdx.x * blockDim.x + threadIdx.x;
    if (b < B_) {
        const float* row = x + (size_t)b * IN_;
        float a = 0.0f;
        for (int p = 0; p < P_; p++) a += row[p] * wn[p];
        s[b] = a;
    }
}

// ---- x (f32) -> xb (bf16), 8 elements per thread ----
__global__ void cvt_x_kernel(const float* __restrict__ x, unsigned short* __restrict__ xb) {
    size_t n8 = (size_t)B_ * IN_ / 8;
    size_t stride = (size_t)gridDim.x * blockDim.x;
    for (size_t i = blockIdx.x * (size_t)blockDim.x + threadIdx.x; i < n8; i += stride) {
        const float4* p = (const float4*)(x + i * 8);
        float4 a = p[0], b = p[1];
        union { uint4 u; unsigned short h[8]; } o;
        o.h[0] = f2bf(a.x); o.h[1] = f2bf(a.y); o.h[2] = f2bf(a.z); o.h[3] = f2bf(a.w);
        o.h[4] = f2bf(b.x); o.h[5] = f2bf(b.y); o.h[6] = f2bf(b.z); o.h[7] = f2bf(b.w);
        ((uint4*)xb)[i] = o.u;
    }
}

// ---- Wb[o,i] = bf16( W[o,i] + (i<100 ? 0.1*PT[o,i] : 0) ), 2 elements/thread ----
__global__ void fold_w_kernel(const float* __restrict__ w, const float* __restrict__ pt,
                              unsigned short* __restrict__ wb) {
    size_t n2 = (size_t)OUT_ * IN_ / 2;
    size_t stride = (size_t)gridDim.x * blockDim.x;
    for (size_t idx = blockIdx.x * (size_t)blockDim.x + threadIdx.x; idx < n2; idx += stride) {
        size_t o = idx / (IN_ / 2);
        int i = (int)(idx % (IN_ / 2)) * 2;
        float v0 = w[o * IN_ + i];
        float v1 = w[o * IN_ + i + 1];
        if (i < P_)     v0 += 0.1f * pt[o * P_ + i];
        if (i + 1 < P_) v1 += 0.1f * pt[o * P_ + i + 1];
        union { unsigned int u; unsigned short h[2]; } pk;
        pk.h[0] = f2bf(v0); pk.h[1] = f2bf(v1);
        ((unsigned int*)wb)[idx] = pk.u;
    }
}

// ---- main bf16 WMMA GEMM, fused epilogue: + bias + s*qs, ReLU ----
// block: 256 threads = 8 wave32; block tile 128(M) x 256(N); wave tile 64x64 = 4x4 WMMA tiles
__global__ __launch_bounds__(256, 1) void gemm_kernel(
    const unsigned short* __restrict__ xb,   // [B_, IN_] bf16 row-major
    const unsigned short* __restrict__ wb,   // [OUT_, IN_] bf16 row-major
    const float* __restrict__ bias,
    const float* __restrict__ s,
    const float* __restrict__ qs,
    float* __restrict__ out)                 // [B_, OUT_] f32
{
    const int lane   = threadIdx.x & 31;
    const int wave   = threadIdx.x >> 5;
    const int lane16 = lane & 15;
    const int hi     = lane >> 4;            // 0 = lanes 0-15, 1 = lanes 16-31

    const int waveM = wave & 1;              // 0..1
    const int waveN = wave >> 1;             // 0..3

    const int bBase = blockIdx.y * 128 + waveM * 64;
    const int oBase = blockIdx.x * 256 + waveN * 64;

    // A fragment (16-bit 16x32): lane = M; low lanes K {0..15}, high lanes K {16..31}
    const unsigned short* aRow[4];
    #pragma unroll
    for (int m = 0; m < 4; m++)
        aRow[m] = xb + (size_t)(bBase + m * 16 + lane16) * IN_;

    // B fragment (16-bit 32x16): lane = N; low lanes K {0..15}, high lanes K {16..31}
    const unsigned short* bRow[4];
    #pragma unroll
    for (int n = 0; n < 4; n++)
        bRow[n] = wb + (size_t)(oBase + n * 16 + lane16) * IN_;

    v8f acc[4][4];
    #pragma unroll
    for (int m = 0; m < 4; m++)
        #pragma unroll
        for (int n = 0; n < 4; n++)
            #pragma unroll
            for (int r = 0; r < 8; r++) acc[m][n][r] = 0.0f;

    for (int k = 0; k < IN_; k += 32) {
        const int klo = k + hi * 16;   // A per-lane K base
        const int kb  = k + hi * 16;   // B per-lane K base
        Frag a[4], b[4];
        #pragma unroll
        for (int m = 0; m < 4; m++) {
            a[m].u[0] = *(const uint4*)(aRow[m] + klo);
            a[m].u[1] = *(const uint4*)(aRow[m] + klo + 8);
        }
        #pragma unroll
        for (int n = 0; n < 4; n++) {
            b[n].u[0] = *(const uint4*)(bRow[n] + kb);
            b[n].u[1] = *(const uint4*)(bRow[n] + kb + 8);
        }
        #pragma unroll
        for (int m = 0; m < 4; m++)
            #pragma unroll
            for (int n = 0; n < 4; n++)
                acc[m][n] = __builtin_amdgcn_wmma_f32_16x16x32_bf16(
                    false, a[m].bf, false, b[n].bf, (short)0, acc[m][n], false, false);
    }

    // Epilogue. C layout: VGPR r, lanes 0-15 -> M=r, lanes 16-31 -> M=r+8; N = lane%16.
    float bq[4], qv[4];
    int col[4];
    #pragma unroll
    for (int n = 0; n < 4; n++) {
        col[n] = oBase + n * 16 + lane16;
        bq[n]  = bias[col[n]];
        qv[n]  = qs[col[n]];
    }
    #pragma unroll
    for (int m = 0; m < 4; m++) {
        const int rowBase = bBase + m * 16 + hi * 8;
        #pragma unroll
        for (int r = 0; r < 8; r++) {
            const int row = rowBase + r;
            const float sv = s[row];
            float* orow = out + (size_t)row * OUT_;
            #pragma unroll
            for (int n = 0; n < 4; n++) {
                float v = acc[m][n][r] + bq[n] + sv * qv[n];
                orow[col[n]] = v > 0.0f ? v : 0.0f;
            }
        }
    }
}

extern "C" void kernel_launch(void* const* d_in, const int* in_sizes, int n_in,
                              void* d_out, int out_size, void* d_ws, size_t ws_size,
                              hipStream_t stream) {
    const float* x    = (const float*)d_in[0];   // [B_, IN_]
    const float* w    = (const float*)d_in[1];   // [OUT_, IN_]
    const float* bias = (const float*)d_in[2];   // [OUT_]
    const float* pt   = (const float*)d_in[3];   // [OUT_, P_]
    const float* qt   = (const float*)d_in[4];   // [OUT_, Q_]
    float* out = (float*)d_out;

    char* ws = (char*)d_ws;
    unsigned short* xb = (unsigned short*)ws;                              // 64 MiB
    unsigned short* wb = (unsigned short*)(ws + (size_t)B_ * IN_ * 2);     // 32 MiB
    float* s  = (float*)(ws + (size_t)B_ * IN_ * 2 + (size_t)OUT_ * IN_ * 2);
    float* qs = s + B_;
    float* wn = qs + OUT_;

    wn_kernel<<<1, 128, 0, stream>>>(wn);
    qs_kernel<<<(OUT_ + 255) / 256, 256, 0, stream>>>(qt, qs);
    s_kernel<<<(B_ + 255) / 256, 256, 0, stream>>>(x, wn, s);
    cvt_x_kernel<<<2048, 256, 0, stream>>>(x, xb);
    fold_w_kernel<<<2048, 256, 0, stream>>>(w, pt, wb);

    dim3 grid(OUT_ / 256, B_ / 128);
    gemm_kernel<<<grid, 256, 0, stream>>>(xb, wb, bias, s, qs, out);
}
